// SkeletalMotionInterpolator_70557722738757
// MI455X (gfx1250) — matrix operand order repaired and checked by reading, not compiled, not run on hardware
//
#include <hip/hip_runtime.h>
#include <hip/hip_bf16.h>

// ---------------- problem constants (from reference) ----------------
static constexpr int kN      = 49152;   // B*J nodes
static constexpr int kHC     = 1024;    // hidden channels (HEADS*HID)
static constexpr int kHID    = 128;
static constexpr int kHEADS  = 8;
static constexpr int kJ      = 24;
static constexpr int kB      = 2048;
static constexpr int kRH     = 512;
static constexpr int kRotOut = 4320;    // J*T_LEN*F_IN
static constexpr int kRootIn = 36;      // CTX*GF
static constexpr int kRootOut= 90;      // T_LEN*GF

typedef __attribute__((ext_vector_type(16))) __bf16        v16bf;
typedef __attribute__((ext_vector_type(8)))  float         v8f;
typedef __attribute__((ext_vector_type(8)))  unsigned short ushort8;

__device__ __forceinline__ unsigned short f2bf(float f) {
  unsigned u = __builtin_bit_cast(unsigned, f);
  u += 0x7FFFu + ((u >> 16) & 1u);          // round-to-nearest-even
  return (unsigned short)(u >> 16);
}
__device__ __forceinline__ float bf2f(unsigned short h) {
  unsigned u = ((unsigned)h) << 16;
  return __builtin_bit_cast(float, u);
}
__device__ __forceinline__ float lrelu(float v, float s) { return v > 0.f ? v : s * v; }

union FragU { v16bf v; ushort8 u[2]; };

// ---------------- bf16 WMMA GEMM: out[M,Nt] = A[M,K] @ Bt[Nt,K]^T ----------------
// Register-tiled: each wave owns a 32x64 output tile (2 M-tiles x 4 N-tiles),
// so each K-step does 6 fragment loads feeding 8 independent WMMAs
// (~21 FLOP per fragment byte, 8-deep independent XDL chain).
// A, Bt bf16 row-major (Bt = weight transposed so K is contiguous).
// f32 accumulation; optional bias / leaky-relu; stores f32 or bf16.
__global__ __launch_bounds__(256)
void gemm_bf16_wmma(const unsigned short* __restrict__ A,
                    const unsigned short* __restrict__ Bt,
                    const float* __restrict__ bias,
                    void* __restrict__ out,
                    int M, int Nt, int K, int out_f32, int act)
{
  const int lane   = threadIdx.x & 31;
  const int wave   = threadIdx.x >> 5;
  const int nTiles = Nt >> 4;
  const int tn0    = (blockIdx.x * 8 + wave) * 4;   // first of 4 N-tiles
  if (tn0 >= nTiles) return;
  const int m0 = blockIdx.y << 5;                   // 32 rows per block-row
  const int lo = lane & 15;
  const int hi = lane >> 4;
  const int aoff = hi << 3;   // A halves: K[hi*8 .. +7] and K[16+hi*8 .. +7]
  const int boff = hi << 3;   // B halves: K[hi*8 .. +7] and K[16+hi*8 .. +7]

  const unsigned short* arow0 = A + (size_t)(m0 + lo) * K;
  const unsigned short* arow1 = A + (size_t)(m0 + 16 + lo) * K;
  const unsigned short* brow[4];
  bool valid[4];
#pragma unroll
  for (int t = 0; t < 4; ++t) {
    const int tn = tn0 + t;
    valid[t] = tn < nTiles;
    const int tcl = valid[t] ? tn : (nTiles - 1);   // clamp address, mask store
    brow[t] = Bt + (size_t)((tcl << 4) + lo) * K;
  }

  v8f acc[2][4] = {};
  for (int kb = 0; kb < K; kb += 32) {
    FragU fa0, fa1, fb[4];
    fa0.u[0] = *(const ushort8*)(arow0 + kb + aoff);
    fa0.u[1] = *(const ushort8*)(arow0 + kb + aoff + 16);
    fa1.u[0] = *(const ushort8*)(arow1 + kb + aoff);
    fa1.u[1] = *(const ushort8*)(arow1 + kb + aoff + 16);
#pragma unroll
    for (int t = 0; t < 4; ++t) {
      fb[t].u[0] = *(const ushort8*)(brow[t] + kb + boff);
      fb[t].u[1] = *(const ushort8*)(brow[t] + kb + boff + 16);
    }
    if (kb + 32 < K) {
      __builtin_prefetch(arow0 + kb + 32 + aoff, 0, 1);   // global_prefetch_b8
      __builtin_prefetch(arow1 + kb + 32 + aoff, 0, 1);
#pragma unroll
      for (int t = 0; t < 4; ++t)
        __builtin_prefetch(brow[t] + kb + 32 + boff, 0, 1);
    }
#pragma unroll
    for (int t = 0; t < 4; ++t) {
      acc[0][t] = __builtin_amdgcn_wmma_f32_16x16x32_bf16(
          false, fa0.v, false, fb[t].v, (short)0, acc[0][t], false, false);
      acc[1][t] = __builtin_amdgcn_wmma_f32_16x16x32_bf16(
          false, fa1.v, false, fb[t].v, (short)0, acc[1][t], false, false);
    }
  }

#pragma unroll
  for (int t = 0; t < 4; ++t) {
    if (!valid[t]) continue;
    const int col = ((tn0 + t) << 4) + lo;
    const float bv = bias ? bias[col] : 0.f;
#pragma unroll
    for (int mt = 0; mt < 2; ++mt) {
#pragma unroll
      for (int r = 0; r < 8; ++r) {
        const int row = m0 + (mt << 4) + r + (hi << 3);
        float v = acc[mt][t][r] + bv;
        if (act) v = lrelu(v, 0.01f);
        if (out_f32) ((float*)out)[(size_t)row * Nt + col] = v;
        else ((unsigned short*)out)[(size_t)row * Nt + col] = f2bf(v);
      }
    }
  }
}

// ---------------- weight convert + transpose: W[K,Nt] f32 -> Wt[Nt,K] bf16 ----------------
__global__ void cvt_transpose_bf16(const float* __restrict__ W, unsigned short* __restrict__ Wt,
                                   int K, int Nt) {
  size_t i = (size_t)blockIdx.x * blockDim.x + threadIdx.x;
  if (i >= (size_t)K * Nt) return;
  int n = (int)(i % Nt);
  int k = (int)(i / Nt);
  Wt[(size_t)n * K + k] = f2bf(W[i]);
}

// ---------------- layer-0 projection: h = x[N,6] @ W0[6,HC] (K too small for WMMA) ----------------
__global__ void layer0_kernel(const float* __restrict__ x, const float* __restrict__ W0,
                              unsigned short* __restrict__ h) {
  size_t i = (size_t)blockIdx.x * blockDim.x + threadIdx.x;
  if (i >= (size_t)kN * kHC) return;
  int c = (int)(i & (kHC - 1));
  size_t n = i >> 10;
  const float* xr = x + n * 6;
  float s = 0.f;
#pragma unroll
  for (int k = 0; k < 6; ++k) s += xr[k] * W0[k * kHC + c];
  h[i] = f2bf(s);
}

// ---------------- attention logits: al_s/al_d[n,head] = <h[n,head,:], att[head,:]> ----------------
__global__ void attn_logits_kernel(const unsigned short* __restrict__ h,
                                   const float* __restrict__ a_s, const float* __restrict__ a_d,
                                   float* __restrict__ al_s, float* __restrict__ al_d) {
  int i = blockIdx.x * blockDim.x + threadIdx.x;
  if (i >= kN * kHEADS) return;
  int head = i & (kHEADS - 1);
  int n = i >> 3;
  const unsigned short* hr = h + (size_t)n * kHC + head * kHID;
  const float* vs = a_s + head * kHID;
  const float* vd = a_d + head * kHID;
  float ss = 0.f, sd = 0.f;
  for (int c = 0; c < kHID; ++c) {
    float f = bf2f(hr[c]);
    ss += f * vs[c];
    sd += f * vd[c];
  }
  al_s[i] = ss;
  al_d[i] = sd;
}

// ---------------- GAT softmax + aggregate (static chain: srcs = {self, j-1, j+1}) ----------------
__global__ __launch_bounds__(256)
void gat_aggregate_kernel(const unsigned short* __restrict__ h,
                          const float* __restrict__ al_s, const float* __restrict__ al_d,
                          const float* __restrict__ bias, unsigned short* __restrict__ out,
                          int act) {
  const int n = blockIdx.x;        // node
  const int t = threadIdx.x;       // 256 threads * 4 channels = 1024
  const int c = t << 2;
  const int head = c >> 7;
  const int j = n % kJ;

  int s[3]; int ns = 0;
  s[ns++] = n;                     // self loop
  if (j > 0)      s[ns++] = n - 1;
  if (j < kJ - 1) s[ns++] = n + 1;

  const float ad = al_d[n * kHEADS + head];
  float e[3];
  float m = -1e30f;
  for (int k = 0; k < ns; ++k) {
    float v = lrelu(al_s[s[k] * kHEADS + head] + ad, 0.2f);
    e[k] = v;
    m = fmaxf(m, v);
  }
  float den = 0.f;
  for (int k = 0; k < ns; ++k) { e[k] = __expf(e[k] - m); den += e[k]; }
  const float inv = 1.f / den;

  float a0 = bias[c], a1 = bias[c + 1], a2 = bias[c + 2], a3 = bias[c + 3];
  for (int k = 0; k < ns; ++k) {
    const float a = e[k] * inv;
    const unsigned short* hr = h + (size_t)s[k] * kHC + c;
    a0 += a * bf2f(hr[0]);
    a1 += a * bf2f(hr[1]);
    a2 += a * bf2f(hr[2]);
    a3 += a * bf2f(hr[3]);
  }
  if (act) {
    a0 = lrelu(a0, 0.01f); a1 = lrelu(a1, 0.01f);
    a2 = lrelu(a2, 0.01f); a3 = lrelu(a3, 0.01f);
  }
  unsigned short* o = out + (size_t)n * kHC + c;
  o[0] = f2bf(a0); o[1] = f2bf(a1); o[2] = f2bf(a2); o[3] = f2bf(a3);
}

// ---------------- global mean pool over J nodes per graph ----------------
__global__ void pool_kernel(const unsigned short* __restrict__ act, unsigned short* __restrict__ g) {
  int i = blockIdx.x * blockDim.x + threadIdx.x;
  if (i >= kB * kHC) return;
  int c = i & (kHC - 1);
  int b = i >> 10;
  float s = 0.f;
  const unsigned short* base = act + (size_t)b * kJ * kHC + c;
  for (int j = 0; j < kJ; ++j) s += bf2f(base[(size_t)j * kHC]);
  g[i] = f2bf(s * (1.f / kJ));
}

// ---------------- tiny f32 MLP layer (root head; negligible FLOPs) ----------------
__global__ void mlp_kernel(const float* __restrict__ in, const float* __restrict__ W,
                           const float* __restrict__ bias, float* __restrict__ out,
                           int M, int K, int Nt, int act) {
  int i = blockIdx.x * blockDim.x + threadIdx.x;
  if (i >= M * Nt) return;
  int o = i % Nt;
  int b = i / Nt;
  const float* ir = in + (size_t)b * K;
  float s = bias[o];
  for (int k = 0; k < K; ++k) s += ir[k] * W[(size_t)k * Nt + o];
  if (act) s = lrelu(s, 0.01f);
  out[i] = s;
}

// ---------------- driver ----------------
extern "C" void kernel_launch(void* const* d_in, const int* in_sizes, int n_in,
                              void* d_out, int out_size, void* d_ws, size_t ws_size,
                              hipStream_t stream) {
  (void)in_sizes; (void)n_in; (void)out_size; (void)ws_size;
  const float* x        = (const float*)d_in[0];
  // d_in[1] edge_index / d_in[2] batch: static chain structure, handled analytically
  const float* root_ctx = (const float*)d_in[3];
  const float* W0       = (const float*)d_in[4];
  const float* Wh       = (const float*)d_in[5];
  const float* att_src  = (const float*)d_in[6];
  const float* att_dst  = (const float*)d_in[7];
  const float* bias_g   = (const float*)d_in[8];
  const float* Wr1      = (const float*)d_in[9];
  const float* br1      = (const float*)d_in[10];
  const float* Wr2      = (const float*)d_in[11];
  const float* br2      = (const float*)d_in[12];
  const float* Wp1      = (const float*)d_in[13];
  const float* bp1      = (const float*)d_in[14];
  const float* Wp2      = (const float*)d_in[15];
  const float* bp2      = (const float*)d_in[16];
  const float* Wp3      = (const float*)d_in[17];
  const float* bp3      = (const float*)d_in[18];
  float* out = (float*)d_out;

  char* ws = (char*)d_ws;
  size_t off = 0;
  auto alloc = [&](size_t bytes) {
    char* p = ws + off;
    off += (bytes + 255) & ~(size_t)255;
    return p;
  };
  unsigned short* h_bf   = (unsigned short*)alloc((size_t)kN * kHC * 2);
  unsigned short* act_bf = (unsigned short*)alloc((size_t)kN * kHC * 2);
  float*          al_s   = (float*)alloc((size_t)kN * kHEADS * 4);
  float*          al_d   = (float*)alloc((size_t)kN * kHEADS * 4);
  unsigned short* Wht    = (unsigned short*)alloc((size_t)3 * kHC * kHC * 2);
  unsigned short* Wr1t   = (unsigned short*)alloc((size_t)kHC * kHC * 2);
  unsigned short* Wr2t   = (unsigned short*)alloc((size_t)kHC * kRotOut * 2);
  unsigned short* g_bf   = (unsigned short*)alloc((size_t)kB * kHC * 2);
  unsigned short* h1_bf  = (unsigned short*)alloc((size_t)kB * kHC * 2);
  float*          r1     = (float*)alloc((size_t)kB * kRH * 4);
  float*          r2     = (float*)alloc((size_t)kB * kRH * 4);

  const int T = 256;

  // weight conversions (bf16, transposed so K is contiguous for B-fragments)
  for (int i = 0; i < 3; ++i)
    cvt_transpose_bf16<<<(kHC * kHC + T - 1) / T, T, 0, stream>>>(
        Wh + (size_t)i * kHC * kHC, Wht + (size_t)i * kHC * kHC, kHC, kHC);
  cvt_transpose_bf16<<<(kHC * kHC + T - 1) / T, T, 0, stream>>>(Wr1, Wr1t, kHC, kHC);
  cvt_transpose_bf16<<<(kHC * kRotOut + T - 1) / T, T, 0, stream>>>(Wr2, Wr2t, kHC, kRotOut);

  // layer 0 projection (K=6)
  layer0_kernel<<<((size_t)kN * kHC + T - 1) / T, T, 0, stream>>>(x, W0, h_bf);

  // GEMM grid helper: each block = 8 waves x (32x64) wave-tiles = 32 rows x 512 cols
  auto gemm_grid = [](int M, int Nt) {
    int nTiles = Nt >> 4;
    return dim3((nTiles + 31) / 32, M >> 5);
  };

  // 4 GAT layers
  for (int L = 0; L < 4; ++L) {
    attn_logits_kernel<<<(kN * kHEADS + T - 1) / T, T, 0, stream>>>(
        h_bf, att_src + L * kHEADS * kHID, att_dst + L * kHEADS * kHID, al_s, al_d);
    gat_aggregate_kernel<<<kN, T, 0, stream>>>(
        h_bf, al_s, al_d, bias_g + L * kHC, act_bf, (L < 3) ? 1 : 0);
    if (L < 3) {
      gemm_bf16_wmma<<<gemm_grid(kN, kHC), T, 0, stream>>>(
          act_bf, Wht + (size_t)L * kHC * kHC, nullptr, h_bf, kN, kHC, kHC,
          /*out_f32=*/0, /*act=*/0);
    }
  }

  // global mean pool
  pool_kernel<<<(kB * kHC + T - 1) / T, T, 0, stream>>>(act_bf, g_bf);

  // rotation head: h1 = lrelu(g @ Wr1 + br1); rot = h1 @ Wr2 + br2 -> d_out[0 .. B*4320)
  gemm_bf16_wmma<<<gemm_grid(kB, kHC), T, 0, stream>>>(
      g_bf, Wr1t, br1, h1_bf, kB, kHC, kHC, 0, 1);
  gemm_bf16_wmma<<<gemm_grid(kB, kRotOut), T, 0, stream>>>(
      h1_bf, Wr2t, br2, out, kB, kRotOut, kHC, 1, 0);

  // root head (f32 throughout, tiny) -> d_out[B*4320 .. B*4320 + B*90)
  mlp_kernel<<<(kB * kRH + T - 1) / T, T, 0, stream>>>(root_ctx, Wp1, bp1, r1, kB, kRootIn, kRH, 1);
  mlp_kernel<<<(kB * kRH + T - 1) / T, T, 0, stream>>>(r1, Wp2, bp2, r2, kB, kRH, kRH, 1);
  mlp_kernel<<<(kB * kRootOut + T - 1) / T, T, 0, stream>>>(
      r2, Wp3, bp3, out + (size_t)kB * kRotOut, kB, kRH, kRootOut, 0);
}
